// GraphRecurrent_39908836115117
// MI455X (gfx1250) — matrix operand order repaired and checked by reading, not compiled, not run on hardware
//
#include <hip/hip_runtime.h>

typedef __attribute__((ext_vector_type(16))) _Float16 v16h;
typedef __attribute__((ext_vector_type(8)))  float    v8f;

#define BSZ   32
#define TT    50
#define NN    192
#define HH    32
#define RT    16
#define NSTEP (TT - 1)
#define BLOCK 128

__device__ __forceinline__ float tanh_fast(float x) {
  float r;
  asm("v_tanh_f32 %0, %1" : "=v"(r) : "v"(x));
  return r;
}
__device__ __forceinline__ float sigmoid_fast(float x) {
  return 1.0f / (1.0f + __expf(-x));
}

__global__ void gkt_zero(float* __restrict__ p, int n) {
  int i = blockIdx.x * blockDim.x + threadIdx.x;
  if (i < n) p[i] = 0.0f;
}

__global__ void __launch_bounds__(BLOCK)
gkt_step(int t,
         const int*   __restrict__ skill,
         const float* __restrict__ adj,
         const float* __restrict__ emb,
         const float* __restrict__ Wmsg1, const float* __restrict__ bmsg1,
         const float* __restrict__ Wmsg2, const float* __restrict__ bmsg2,
         const float* __restrict__ Whr,  const float* __restrict__ Whi,
         const float* __restrict__ Whh,
         const float* __restrict__ Wir,  const float* __restrict__ bir,
         const float* __restrict__ Wii,  const float* __restrict__ bii,
         const float* __restrict__ Win,  const float* __restrict__ bin_,
         const float* __restrict__ Wo1,  const float* __restrict__ bo1,
         const float* __restrict__ Wo2,  const float* __restrict__ bo2,
         const float* __restrict__ Wo3,  const float* __restrict__ bo3,
         float* __restrict__ hid, float* __restrict__ out)
{
  extern __shared__ float smem[];
  float* s_send = smem;               // NN*HH = 6144
  float* s_hid  = s_send + NN*HH;     // 6144  (also absorbs s_send prefetch overrun)
  float* s_adjT = s_hid  + NN*HH;     // NN*RT = 3072   [s][j] = adj[b, r0+j, s]
  float* s_rb   = s_adjT + NN*RT;     // RT*HH = 512    (absorbs s_adjT prefetch overrun)
  float* s_agg  = s_rb   + RT*HH;     // 512
  float* s_hn   = s_agg  + RT*HH;     // 512
  float* s_p    = s_hn   + RT*HH;     // 512
  float* s_xg   = s_p    + RT*HH;     // 96 (+pad)

  const int tid  = threadIdx.x;
  const int lane = tid & 31;
  const int wave = tid >> 5;
  const int b    = blockIdx.x / (NN / RT);
  const int r0   = (blockIdx.x % (NN / RT)) * RT;

  // ---------- phase 0: zero agg, stage hidden[b] and adj^T tile ----------
  for (int e = tid; e < RT*HH; e += BLOCK) s_agg[e] = 0.0f;
  {
    const float4* hg4 = (const float4*)(hid + (size_t)b * NN * HH);
    float4* sh4 = (float4*)s_hid;
    for (int e = tid; e < NN*HH/4; e += BLOCK) sh4[e] = hg4[e];
  }
  for (int e = tid; e < NN*RT; e += BLOCK) {
    int j = e / NN, s = e - j * NN;
    s_adjT[s*RT + j] = adj[((size_t)b * NN + (r0 + j)) * NN + s];
  }
  __syncthreads();

  // ---------- phase 1: send_p for all 192 rows, rb for 16 r rows ----------
  {
    const int k = tid & 31;                       // fixed output channel per thread
    float ws[32], wr[32];
    #pragma unroll
    for (int j = 0; j < 32; ++j) {
      ws[j] = Wmsg1[k * 2*HH + j];                // W_send row k
      wr[j] = Wmsg1[k * 2*HH + HH + j];           // W_recv row k
    }
    for (int e = tid; e < NN*HH; e += BLOCK) {    // e & 31 == k always
      int s = e >> 5;
      const float* hrow = s_hid + s * HH;
      float acc = 0.0f;
      #pragma unroll
      for (int j = 0; j < 32; ++j) acc += hrow[j] * ws[j];
      s_send[e] = acc;
    }
    const float bmk = bmsg1[k];
    for (int e = tid; e < RT*HH; e += BLOCK) {
      int r = e >> 5;
      const float* hrow = s_hid + (r0 + r) * HH;
      float acc = bmk;
      #pragma unroll
      for (int j = 0; j < 32; ++j) acc += hrow[j] * wr[j];
      s_rb[e] = acc;
    }
  }
  __syncthreads();

  // ---------- phase 2: fused msg / msg2 / agg via WMMA (software pipelined) ----------
  // A-fragment per-lane layout (16-bit A 16x32): lane<16 -> K 0..7,16..23 ; lane>=16 -> K 8..15,24..31
  const int m      = lane & 15;
  const int hiHalf = lane >> 4;
  const int kb1    = hiHalf ? 8 : 0;
  const int kb2    = kb1 + 16;
  const int kB     = hiHalf ? 16 : 0;   // unused; B-fragment K follows kb1/kb2 like A
  const int j0     = hiHalf ? 8 : 0;    // C/D: M = j0 + j

  float rbv[16];
  #pragma unroll
  for (int i = 0; i < 8; ++i) {
    rbv[i]     = s_rb[m*HH + kb1 + i];
    rbv[i + 8] = s_rb[m*HH + kb2 + i];
  }
  // B = W_msg2^T : B[k][n] = Wmsg2[n*HH + k]; per-lane n = lane&15, K = kb1..kb1+7, kb2..kb2+7
  v16h bmat0, bmat1;
  #pragma unroll
  for (int i = 0; i < 16; ++i) {
    bmat0[i] = (_Float16)Wmsg2[m*HH + kb1 + i + (i & 8)];          // n-half 0
    bmat1[i] = (_Float16)Wmsg2[(m + 16)*HH + kb1 + i + (i & 8)];   // n-half 1
  }
  const float bm2a = bmsg2[m];
  const float bm2b = bmsg2[m + 16];

  // structurally 16B-aligned LDS access (ds_load_b128)
  const float4* sb4 = (const float4*)s_send;    // 8 float4 per send row
  const float4* ab4 = (const float4*)s_adjT;    // 4 float4 per adjT row
  const int sk4 = kb1 >> 2;                     // 0 or 2
  const int ak4 = j0  >> 2;                     // 0 or 2

  float acc0[8] = {}; float acc1[8] = {};
  v8f zero8 = {0.f,0.f,0.f,0.f,0.f,0.f,0.f,0.f};

  // prologue: build A-fragment for first s
  v16h a_cur;
  float ajc[8];
  {
    const int s = wave;
    float4 c0 = sb4[s*8 + sk4];
    float4 c1 = sb4[s*8 + sk4 + 1];
    float4 c2 = sb4[s*8 + sk4 + 4];
    float4 c3 = sb4[s*8 + sk4 + 5];
    float sv[16] = {c0.x,c0.y,c0.z,c0.w, c1.x,c1.y,c1.z,c1.w,
                    c2.x,c2.y,c2.z,c2.w, c3.x,c3.y,c3.z,c3.w};
    #pragma unroll
    for (int i = 0; i < 16; ++i)
      a_cur[i] = (_Float16)tanh_fast(rbv[i] + sv[i]);
    float4 a0 = ab4[s*4 + ak4];
    float4 a1 = ab4[s*4 + ak4 + 1];
    ajc[0]=a0.x; ajc[1]=a0.y; ajc[2]=a0.z; ajc[3]=a0.w;
    ajc[4]=a1.x; ajc[5]=a1.y; ajc[6]=a1.z; ajc[7]=a1.w;
  }

  // scalar loop counter: compile-time trip count -> pure-SALU loop control,
  // no divergent exec-mask bookkeeping; EXEC stays all-ones for WMMA.
  #pragma unroll 2
  for (int it = 0; it < NN/4; ++it) {
    v8f d0 = __builtin_amdgcn_wmma_f32_16x16x32_f16(false, a_cur, false, bmat0,
                                                    (short)0, zero8, false, false);
    v8f d1 = __builtin_amdgcn_wmma_f32_16x16x32_f16(false, a_cur, false, bmat1,
                                                    (short)0, zero8, false, false);
    // build next A-fragment while WMMAs are in flight. Final-iteration
    // overreads land in adjacent LDS buffers (in-bounds, results unused).
    const int sn = (it + 1) * 4 + wave;
    float4 c0 = sb4[sn*8 + sk4];
    float4 c1 = sb4[sn*8 + sk4 + 1];
    float4 c2 = sb4[sn*8 + sk4 + 4];
    float4 c3 = sb4[sn*8 + sk4 + 5];
    float sv[16] = {c0.x,c0.y,c0.z,c0.w, c1.x,c1.y,c1.z,c1.w,
                    c2.x,c2.y,c2.z,c2.w, c3.x,c3.y,c3.z,c3.w};
    v16h a_nxt;
    #pragma unroll
    for (int i = 0; i < 16; ++i)
      a_nxt[i] = (_Float16)tanh_fast(rbv[i] + sv[i]);
    float4 a0 = ab4[sn*4 + ak4];
    float4 a1 = ab4[sn*4 + ak4 + 1];
    float ajn[8] = {a0.x,a0.y,a0.z,a0.w, a1.x,a1.y,a1.z,a1.w};

    // consume D = msg2 pre-activation: bias + tanh + adj-weighted accumulate
    #pragma unroll
    for (int j = 0; j < 8; ++j) {
      acc0[j] += ajc[j] * tanh_fast(d0[j] + bm2a);
      acc1[j] += ajc[j] * tanh_fast(d1[j] + bm2b);
    }
    a_cur = a_nxt;
    #pragma unroll
    for (int j = 0; j < 8; ++j) ajc[j] = ajn[j];
  }
  #pragma unroll
  for (int j = 0; j < 8; ++j) {
    atomicAdd(&s_agg[(j0 + j)*HH + m],       acc0[j]);
    atomicAdd(&s_agg[(j0 + j)*HH + m + 16],  acc1[j]);
  }

  // ---------- phase 3: input gates (r-independent) ----------
  if (tid < 3*HH) {
    int g = tid >> 5, hq = tid & 31;
    const float* W  = (g == 0) ? Wir : (g == 1) ? Wii : Win;
    const float* bb = (g == 0) ? bir : (g == 1) ? bii : bin_;
    int sk = skill[b * TT + t];
    const float* x = emb + sk * HH;
    float acc = bb[hq];
    #pragma unroll
    for (int j = 0; j < 32; ++j) acc += x[j] * W[hq*HH + j];
    s_xg[tid] = acc;
  }
  __syncthreads();

  // ---------- phase 4: GRU update ----------
  const float inv_n = 1.0f / (float)NN;
  for (int e = tid; e < RT*HH; e += BLOCK) {
    int r = e >> 5, hq = e & 31;
    const float* ag = s_agg + r * HH;
    float ar = 0.f, ai = 0.f, ah = 0.f;
    #pragma unroll
    for (int j = 0; j < 32; ++j) {
      float av = ag[j] * inv_n;
      ar += av * Whr[hq*HH + j];
      ai += av * Whi[hq*HH + j];
      ah += av * Whh[hq*HH + j];
    }
    float rg = sigmoid_fast(s_xg[hq] + ar);
    float ig = sigmoid_fast(s_xg[HH + hq] + ai);
    float ng = tanh_fast(s_xg[2*HH + hq] + rg * ah);
    float hp = s_hid[(r0 + r)*HH + hq];
    float hn = (1.0f - ig) * ng + ig * hp;
    s_hn[e] = hn;
    hid[((size_t)b * NN + r0 + r)*HH + hq] = hn;
  }
  __syncthreads();

  // ---------- phase 5: output MLP ----------
  for (int e = tid; e < RT*HH; e += BLOCK) {
    int r = e >> 5, hq = e & 31;
    float acc = bo1[hq];
    #pragma unroll
    for (int j = 0; j < 32; ++j) acc += s_hn[r*HH + j] * Wo1[hq*HH + j];
    s_p[e] = fmaxf(acc, 0.0f);
  }
  __syncthreads();
  for (int e = tid; e < RT*HH; e += BLOCK) {
    int r = e >> 5, hq = e & 31;
    float acc = bo2[hq];
    #pragma unroll
    for (int j = 0; j < 32; ++j) acc += s_p[r*HH + j] * Wo2[hq*HH + j];
    s_agg[e] = fmaxf(acc, 0.0f);        // reuse s_agg as p2 buffer
  }
  __syncthreads();
  for (int e = tid; e < RT*HH; e += BLOCK) {
    int r = e >> 5, hq = e & 31;
    float acc = bo3[hq];
    #pragma unroll
    for (int j = 0; j < 32; ++j) acc += s_agg[r*HH + j] * Wo3[hq*HH + j];
    out[(((size_t)b * NSTEP + t) * NN + r0 + r)*HH + hq] = acc;
  }
}

extern "C" void kernel_launch(void* const* d_in, const int* in_sizes, int n_in,
                              void* d_out, int out_size, void* d_ws, size_t ws_size,
                              hipStream_t stream) {
  const int*   skill = (const int*)  d_in[0];
  const float* adj   = (const float*)d_in[1];
  const float* emb   = (const float*)d_in[2];
  const float* Wmsg1 = (const float*)d_in[3];
  const float* bmsg1 = (const float*)d_in[4];
  const float* Wmsg2 = (const float*)d_in[5];
  const float* bmsg2 = (const float*)d_in[6];
  const float* Whr   = (const float*)d_in[7];
  const float* Whi   = (const float*)d_in[8];
  const float* Whh   = (const float*)d_in[9];
  const float* Wir   = (const float*)d_in[10];
  const float* bir   = (const float*)d_in[11];
  const float* Wii   = (const float*)d_in[12];
  const float* bii   = (const float*)d_in[13];
  const float* Win   = (const float*)d_in[14];
  const float* bin_  = (const float*)d_in[15];
  const float* Wo1   = (const float*)d_in[16];
  const float* bo1   = (const float*)d_in[17];
  const float* Wo2   = (const float*)d_in[18];
  const float* bo2   = (const float*)d_in[19];
  const float* Wo3   = (const float*)d_in[20];
  const float* bo3   = (const float*)d_in[21];

  float* hid = (float*)d_ws;                 // BSZ*NN*HH f32 recurrent state
  float* out = (float*)d_out;

  const int hid_elems = BSZ * NN * HH;
  gkt_zero<<<(hid_elems + 255) / 256, 256, 0, stream>>>(hid, hid_elems);

  const int grid = BSZ * (NN / RT);          // 384 workgroups
  const size_t smem_bytes =
      (size_t)(NN*HH + NN*HH + NN*RT + 4*RT*HH + 128) * sizeof(float); // ~70 KB

  for (int t = 0; t < NSTEP; ++t) {
    gkt_step<<<grid, BLOCK, smem_bytes, stream>>>(
        t, skill, adj, emb,
        Wmsg1, bmsg1, Wmsg2, bmsg2,
        Whr, Whi, Whh,
        Wir, bir, Wii, bii, Win, bin_,
        Wo1, bo1, Wo2, bo2, Wo3, bo3,
        hid, out);
  }
}
